// RNN_speed_test_19670950216562
// MI455X (gfx1250) — matrix operand order repaired and checked by reading, not compiled, not run on hardware
//
#include <hip/hip_runtime.h>
#include <hip/hip_bf16.h>

#define B_   32
#define T_   2048
#define I_   512
#define H_   512
#define G4_  2048      // 4*H
#define KCAT 1024      // I + H combined K
#define NWG  32
#define WGT  256

typedef __attribute__((ext_vector_type(16))) __bf16         v16bf;
typedef __attribute__((ext_vector_type(8)))  float          v8f;
typedef __attribute__((ext_vector_type(4)))  unsigned int   u32x4;
typedef __attribute__((ext_vector_type(4)))  float          f32x4;
typedef __attribute__((ext_vector_type(4)))  unsigned short u16x4;

union Frag {
    v16bf          v;
    u32x4          q[2];
    unsigned short us[16];
    __bf16         h[16];
};

__device__ __forceinline__ unsigned short f2bf(float f) {
    unsigned int u = __float_as_uint(f);
    unsigned int r = u + 0x7FFFu + ((u >> 16) & 1u);   // round-to-nearest-even
    return (unsigned short)(r >> 16);
}

// ---------------- init: zero h ping-pong buffers + grid-barrier counter ----
__global__ void lstm_init_kernel(unsigned short* hbuf, unsigned int* cnt) {
    int idx = blockIdx.x * blockDim.x + threadIdx.x;
    for (int i = idx; i < 2 * B_ * H_; i += gridDim.x * blockDim.x) hbuf[i] = 0;
    if (idx == 0) *cnt = 0;
}

// ---------------- prep: pack [W_ih ; W_hh] -> bf16 Wcat[G4][KCAT] ----------
__global__ void lstm_prep_kernel(const float* __restrict__ Wih,
                                 const float* __restrict__ Whh,
                                 unsigned short* __restrict__ Wcat) {
    int idx = blockIdx.x * blockDim.x + threadIdx.x;   // 0 .. G4*KCAT-1
    int g = idx >> 10;
    int k = idx & 1023;
    float v = (k < I_) ? Wih[g * I_ + k] : Whh[g * H_ + (k - I_)];
    Wcat[idx] = f2bf(v);
}

// ---------------- parallel f32 -> bf16 conversion of the input tensor ------
__global__ void lstm_convx_kernel(const float* __restrict__ x,
                                  unsigned short* __restrict__ xbf) {
    size_t i4 = ((size_t)blockIdx.x * blockDim.x + threadIdx.x) * 4;
    f32x4 v = *(const f32x4*)(x + i4);
    u16x4 o;
    #pragma unroll
    for (int i = 0; i < 4; ++i) o[i] = f2bf(v[i]);
    *(u16x4*)(xbf + i4) = o;
}

// ---------------- persistent recurrence kernel -----------------------------
template <bool PRECONV>
__global__ __launch_bounds__(WGT) void lstm_rec_kernel(
    const float*          __restrict__ x,     // [B][T][I] f32
    const unsigned short* __restrict__ xbf,   // [B][T][I] bf16 (PRECONV only)
    const unsigned short* __restrict__ Wcat,  // [G4][KCAT] bf16
    const float*          __restrict__ bih,
    const float*          __restrict__ bhh,
    float*                __restrict__ out,   // [B][T][H] f32
    unsigned short*       __restrict__ hbuf,  // [2][B][H] bf16
    unsigned int*         __restrict__ cnt)
{
    __shared__ float act[4 * B_ * 16];   // [gate][b][jj]  8 KB
    __shared__ float cst[B_ * 16];       // cell state slice  2 KB

    const int tid  = threadIdx.x;
    const int lane = tid & 31;
    const int w    = tid >> 5;           // wave 0..7
    const int mt   = w & 1;              // M tile (batch 0-15 / 16-31)
    const int gate = w >> 1;             // 0..3 (i,f,g,o)
    const int j0   = blockIdx.x * 16;    // hidden-dim slice base
    const int m0   = mt * 16;
    const int ln   = lane & 15;
    const int lh   = lane >> 4;

    const int   gcol = gate * H_ + j0 + ln;        // column in 4H space
    const float bias = bih[gcol] + bhh[gcol];

    const int arow  = m0 + ln;                     // batch row for A frags
    const int klane = lh * 8;                      // A K sub-offset
    const int krow  = lh * 8;                      // B K sub-offset

    for (int i = tid; i < B_ * 16; i += WGT) cst[i] = 0.f;
    __syncthreads();

    const unsigned short* wbase = Wcat + (size_t)gcol * KCAT;

    // ---- hoist all t-invariant B (weight) fragments into VGPRs ----
    // 32 frags * 32 B/lane = 256 VGPRs/lane (gfx1250 has a 1024-VGPR file)
    Frag bfrag[KCAT / 32];
    #pragma unroll
    for (int kc = 0; kc < KCAT / 32; ++kc) {
        const u32x4* wp = (const u32x4*)(wbase + kc * 32 + krow);
        bfrag[kc].q[0] = wp[0];
        bfrag[kc].q[1] = wp[2];
    }

    const float*          xrowbase  = x   + (size_t)arow * ((size_t)T_ * I_);
    const unsigned short* xbrowbase = xbf + (size_t)arow * ((size_t)T_ * I_);

    unsigned int target = 0;

    for (int t = 0; t < T_; ++t) {
        v8f acc;
        #pragma unroll
        for (int i = 0; i < 8; ++i) acc[i] = bias;   // bias pre-loaded into C

        const unsigned short* hprev = hbuf + (size_t)(t & 1) * (B_ * H_)
                                           + (size_t)arow * H_;

        // ---- K = 0..511 : x_t ----
        if (PRECONV) {
            const unsigned short* xr = xbrowbase + (size_t)t * I_;
            #pragma unroll
            for (int kk = 0; kk < I_; kk += 32) {
                Frag a;
                a.q[0] = *(const u32x4*)(xr + kk + klane);
                a.q[1] = *(const u32x4*)(xr + kk + klane + 16);
                acc = __builtin_amdgcn_wmma_f32_16x16x32_bf16(
                          false, a.v, false, bfrag[kk >> 5].v, (short)0, acc, false, false);
            }
        } else {
            const float* xr = xrowbase + (size_t)t * I_;
            #pragma unroll
            for (int kk = 0; kk < I_; kk += 32) {
                Frag a;
                const f32x4* p0 = (const f32x4*)(xr + kk + klane);
                const f32x4* p1 = (const f32x4*)(xr + kk + klane + 16);
                f32x4 c0 = p0[0], c1 = p0[1], c2 = p1[0], c3 = p1[1];
                #pragma unroll
                for (int i = 0; i < 4; ++i) {
                    a.h[i]      = (__bf16)c0[i];
                    a.h[4 + i]  = (__bf16)c1[i];
                    a.h[8 + i]  = (__bf16)c2[i];
                    a.h[12 + i] = (__bf16)c3[i];
                }
                acc = __builtin_amdgcn_wmma_f32_16x16x32_bf16(
                          false, a.v, false, bfrag[kk >> 5].v, (short)0, acc, false, false);
            }
        }
        // ---- K = 512..1023 : h_{t-1} (bf16) ----
        #pragma unroll
        for (int kk = 0; kk < H_; kk += 32) {
            Frag a;
            a.q[0] = *(const u32x4*)(hprev + kk + klane);
            a.q[1] = *(const u32x4*)(hprev + kk + klane + 16);
            acc = __builtin_amdgcn_wmma_f32_16x16x32_bf16(
                      false, a.v, false, bfrag[16 + (kk >> 5)].v, (short)0, acc, false, false);
        }

        // ---- activations -> LDS ----
        #pragma unroll
        for (int r = 0; r < 8; ++r) {
            float v = acc[r];
            v = (gate == 2) ? tanhf(v) : 1.f / (1.f + __expf(-v));
            int brow = m0 + r + lh * 8;              // C layout: M = r + 8*(lane>>4)
            act[gate * (B_ * 16) + brow * 16 + ln] = v;
        }
        __syncthreads();

        // ---- elementwise c/h update: thread -> (b, jj) pairs ----
        {
            int jj = tid & 15;
            unsigned short* hnext = hbuf + (size_t)((t + 1) & 1) * (B_ * H_);
            #pragma unroll
            for (int rep = 0; rep < 2; ++rep) {
                int b = (tid >> 4) + rep * 16;
                float ig = act[0 * (B_ * 16) + b * 16 + jj];
                float fg = act[1 * (B_ * 16) + b * 16 + jj];
                float gg = act[2 * (B_ * 16) + b * 16 + jj];
                float og = act[3 * (B_ * 16) + b * 16 + jj];
                float c  = fg * cst[b * 16 + jj] + ig * gg;
                float h  = og * tanhf(c);
                cst[b * 16 + jj] = c;
                out[(size_t)b * ((size_t)T_ * H_) + (size_t)t * H_ + j0 + jj] = h;
                hnext[b * H_ + j0 + jj] = f2bf(h);
            }
        }

        // ---- prefetch next timestep's x rows (latency hides under barrier) ----
        if (t + 1 < T_) {
            if (PRECONV) {
                const char* xn = (const char*)(xbrowbase + (size_t)(t + 1) * I_) + lh * 128;
                #pragma unroll
                for (int p = 0; p < 4; ++p)            // 2 lanes/row * 4 = 8 x 128B lines = 1KB row
                    __builtin_prefetch(xn + p * 256, 0, 3);
            } else {
                const char* xn = (const char*)(xrowbase + (size_t)(t + 1) * I_) + lh * 128;
                #pragma unroll
                for (int p = 0; p < 8; ++p)            // 2 lanes/row * 8 = 16 x 128B lines = 2KB row
                    __builtin_prefetch(xn + p * 256, 0, 3);
            }
        }

        // ---- grid barrier across the 32 persistent workgroups ----
        __builtin_amdgcn_fence(__ATOMIC_RELEASE, "agent");
        __syncthreads();
        if (tid == 0) {
            __hip_atomic_fetch_add(cnt, 1u, __ATOMIC_RELAXED, __HIP_MEMORY_SCOPE_AGENT);
            target += NWG;
            while (__hip_atomic_load(cnt, __ATOMIC_RELAXED, __HIP_MEMORY_SCOPE_AGENT) < target) {
                __builtin_amdgcn_s_sleep(1);
            }
        }
        __syncthreads();
        __builtin_amdgcn_fence(__ATOMIC_ACQUIRE, "agent");
    }
}

extern "C" void kernel_launch(void* const* d_in, const int* in_sizes, int n_in,
                              void* d_out, int out_size, void* d_ws, size_t ws_size,
                              hipStream_t stream) {
    const float* x   = (const float*)d_in[0];   // [B,T,I]
    const float* Wih = (const float*)d_in[1];   // [4H,I]
    const float* Whh = (const float*)d_in[2];   // [4H,H]
    const float* bih = (const float*)d_in[3];   // [4H]
    const float* bhh = (const float*)d_in[4];   // [4H]
    float* out = (float*)d_out;                 // [B,T,H]

    // ws layout (bytes):
    //   [0, 1024)            grid-barrier counter
    //   [1024, +64KB)        h ping-pong (bf16)
    //   [66560, +4MB)        packed bf16 Wcat
    //   [4260864, +64MB)     bf16 copy of inputs (optional, if ws large enough)
    const size_t HB_OFF   = 1024;
    const size_t WC_OFF   = HB_OFF + 2ull * B_ * H_ * sizeof(unsigned short);
    const size_t XBF_OFF  = WC_OFF + (size_t)G4_ * KCAT * sizeof(unsigned short);
    const size_t XBF_SIZE = (size_t)B_ * T_ * I_ * sizeof(unsigned short);

    char* ws = (char*)d_ws;
    unsigned int*   cnt  = (unsigned int*)ws;
    unsigned short* hbuf = (unsigned short*)(ws + HB_OFF);
    unsigned short* Wcat = (unsigned short*)(ws + WC_OFF);
    unsigned short* xbf  = (unsigned short*)(ws + XBF_OFF);

    const bool preconv = ws_size >= XBF_OFF + XBF_SIZE;

    lstm_init_kernel<<<32, 256, 0, stream>>>(hbuf, cnt);
    lstm_prep_kernel<<<(G4_ * KCAT) / 256, 256, 0, stream>>>(Wih, Whh, Wcat);
    if (preconv) {
        lstm_convx_kernel<<<((size_t)B_ * T_ * I_ / 4) / 256, 256, 0, stream>>>(x, xbf);
        lstm_rec_kernel<true><<<NWG, WGT, 0, stream>>>(x, xbf, Wcat, bih, bhh, out, hbuf, cnt);
    } else {
        lstm_rec_kernel<false><<<NWG, WGT, 0, stream>>>(x, xbf, Wcat, bih, bhh, out, hbuf, cnt);
    }
}
